// PerformerProjection_53240414601629
// MI455X (gfx1250) — compile-verified, matrix operand order checked by reading, not run on hardware
//
#include <hip/hip_runtime.h>

// ---------------------------------------------------------------------------
// Performer linear attention for MI455X (gfx1250, wave32, WMMA f16->f32,
// async global->LDS staging, double-buffered pass-1 pipeline)
//   B=4 H=8 L=8192 d=64 R=266 (padded to 288 = 18 WMMA tiles)
// ---------------------------------------------------------------------------

typedef __attribute__((ext_vector_type(16))) _Float16 v16h;
typedef __attribute__((ext_vector_type(8)))  float    v8f;
typedef __attribute__((ext_vector_type(4)))  float    f4;

#define EPSF 1e-3f

constexpr int D      = 64;
constexpr int RNUM   = 266;
constexpr int RP     = 288;      // padded R: 18 tiles of 16, 9 chunks of 32
constexpr int RC_N   = RP / 32;  // 9
constexpr int LSEQ   = 8192;
constexpr int HEADS  = 32;       // B*H
constexpr int T1     = 128;      // tokens per sub-chunk, pass 1
constexpr int SUBS   = 4;        // sub-chunks per block, pass 1 (512 tokens)
constexpr int T2     = 128;      // tokens per block, pass 2

// f16 fragment = two contiguous 8-half runs at khoff and khoff+16
static __device__ inline v16h frag_ld(const _Float16* p) {
    v16h r;
    ((f4*)&r)[0] = *(const f4*)p;
    ((f4*)&r)[1] = *(const f4*)(p + 16);
    return r;
}

static __device__ inline v8f vzero8() {
    v8f z = {0.f, 0.f, 0.f, 0.f, 0.f, 0.f, 0.f, 0.f};
    return z;
}

static __device__ inline v8f wmma16(v16h a, v16h b, v8f c) {
    return __builtin_amdgcn_wmma_f32_16x16x32_f16(false, a, false, b,
                                                  (short)0, c, false, false);
}

// CDNA5 async copy: 16B global -> LDS, tracked by ASYNCcnt (no VGPR data).
static __device__ inline void async_b128(const float* g, const float* lds) {
    unsigned l = (unsigned)(unsigned long long)lds;   // addr[31:0] == LDS addr
    asm volatile("global_load_async_to_lds_b128 %0, %1, off"
                 :: "v"(l), "v"(g) : "memory");
}
static __device__ inline void wait_async0() {
    asm volatile("s_wait_asynccnt 0x0" ::: "memory");
}

// ---------------------------------------------------------------------------
// Kernel 0: proj f32 -> f16 padded to [RP][D]; zero kv32 + ksum accumulators.
// ---------------------------------------------------------------------------
__global__ __launch_bounds__(256) void perf_init(const float* __restrict__ proj,
                                                 _Float16* __restrict__ proj16,
                                                 float* __restrict__ ksum,
                                                 float* __restrict__ kv32) {
    const int idx = blockIdx.x * 256 + threadIdx.x;
    if (idx < HEADS * RP * D) kv32[idx] = 0.f;
    if (idx < RP * D) {
        const int r = idx / D, c = idx % D;
        proj16[idx] = (r < RNUM) ? (_Float16)proj[r * D + c] : (_Float16)0.f;
    }
    if (idx < HEADS * RP) ksum[idx] = 0.f;
}

// ---------------------------------------------------------------------------
// Kernel 1: kp = relu(k @ proj^T)+eps ; kv += kp^T @ v ; ksum += sum_t kp
// grid (LSEQ/(T1*SUBS), HEADS) = (16, 32), 256 threads (8 waves)
// Dynamic LDS: kstage[2][T1*D] f32, vstage[2][T1*D] f32, kpT[32][136] f16,
//              ksum_s[RP] f32  -> 140,928 bytes
// ---------------------------------------------------------------------------
__global__ __launch_bounds__(256) void perf_pass1(const float* __restrict__ K,
                                                  const float* __restrict__ V,
                                                  const _Float16* __restrict__ proj16,
                                                  float* __restrict__ ksum,
                                                  float* __restrict__ kv32) {
    extern __shared__ __attribute__((aligned(16))) char smem[];
    float*    kbuf0  = (float*)smem;
    float*    kbuf1  = kbuf0 + T1 * D;
    float*    vbuf0  = kbuf1 + T1 * D;
    float*    vbuf1  = vbuf0 + T1 * D;
    _Float16* kpT    = (_Float16*)(vbuf1 + T1 * D);      // [32][136]
    float*    ksum_s = (float*)(kpT + 32 * 136);

    const int tid   = threadIdx.x;
    const int wave  = tid >> 5;
    const int lane  = tid & 31;
    const int lo    = lane & 15;
    const int khoff = (lane >> 4) * 8;
    const int head  = blockIdx.y;
    const int tok_base = blockIdx.x * (T1 * SUBS);
    const float* Kh = K + (size_t)head * LSEQ * D;
    const float* Vh = V + (size_t)head * LSEQ * D;

    if (wave == 0) __builtin_prefetch(proj16 + (size_t)lane * 576, 0, 3);

    for (int i = tid; i < RP; i += 256) ksum_s[i] = 0.f;

    v8f kvacc[RC_N];
    #pragma unroll
    for (int r = 0; r < RC_N; ++r) kvacc[r] = vzero8();

    // ---- prologue: async-stage tile 0 --------------------------------------
    for (int i = tid; i < T1 * D / 4; i += 256) {
        async_b128(Kh + (size_t)tok_base * D + i * 4, kbuf0 + i * 4);
        async_b128(Vh + (size_t)tok_base * D + i * 4, vbuf0 + i * 4);
    }
    wait_async0();
    __syncthreads();

    for (int ts = 0; ts < SUBS; ++ts) {
        const float* kf = (ts & 1) ? kbuf1 : kbuf0;
        const float* vf = (ts & 1) ? vbuf1 : vbuf0;
        float* kn = (ts & 1) ? kbuf0 : kbuf1;
        float* vn = (ts & 1) ? vbuf0 : vbuf1;

        // ---- prefetch next tile asynchronously while computing -------------
        if (ts + 1 < SUBS) {
            const size_t g0 = (size_t)(tok_base + (ts + 1) * T1) * D;
            for (int i = tid; i < T1 * D / 4; i += 256) {
                async_b128(Kh + g0 + i * 4, kn + i * 4);
                async_b128(Vh + g0 + i * 4, vn + i * 4);
            }
        }

        // ---- convert loop-invariant fragments f32(LDS) -> f16 regs ---------
        v16h akf[2];                       // A: k rows (M-tile = wave), kk=0..1
        {
            const float* row = kf + (size_t)(wave * 16 + lo) * D;
            #pragma unroll
            for (int kk = 0; kk < 2; ++kk) {
                v16h t;
                #pragma unroll
                for (int e = 0; e < 8; ++e) {
                    t[e]     = (_Float16)row[kk * 32 + khoff + e];
                    t[e + 8] = (_Float16)row[kk * 32 + khoff + 16 + e];
                }
                akf[kk] = t;
            }
        }
        v16h bvf[4];                       // B: v^T col (N-tile = wave&3), kk=0..3
        {
            const int n = (wave & 3) * 16 + lo;
            #pragma unroll
            for (int kk = 0; kk < 4; ++kk) {
                v16h t;
                #pragma unroll
                for (int e = 0; e < 8; ++e) {
                    t[e]     = (_Float16)vf[(size_t)(kk * 32 + khoff + e) * D + n];
                    t[e + 8] = (_Float16)vf[(size_t)(kk * 32 + khoff + 16 + e) * D + n];
                }
                bvf[kk] = t;
            }
        }

        for (int rc = 0; rc < RC_N; ++rc) {
            // ---- projection: wave w -> M-tile w, two R sub-tiles -----------
            #pragma unroll
            for (int sub = 0; sub < 2; ++sub) {
                const int rt = rc * 2 + sub;
                v8f acc = vzero8();
                #pragma unroll
                for (int kk = 0; kk < 2; ++kk) {
                    v16h b = frag_ld(proj16 + (size_t)(rt * 16 + lo) * D +
                                     kk * 32 + khoff);
                    acc = wmma16(akf[kk], b, acc);
                }
                const int   rglob  = rt * 16 + lo;          // uniform per lane
                const float addeps = (rglob < RNUM) ? EPSF : 0.f;
                __attribute__((aligned(16))) _Float16 tmp[8];
                #pragma unroll
                for (int e = 0; e < 8; ++e) {
                    float x = acc[e];
                    x = (x > 0.f) ? x : 0.f;                // relu
                    x += addeps;                            // +eps (0 if padded)
                    tmp[e] = (_Float16)x;
                }
                // D-fragment = 8 consecutive M rows -> one b128 transposed store
                *(f4*)&kpT[(size_t)(sub * 16 + lo) * 136 + wave * 16 + khoff] =
                    *(const f4*)tmp;
            }
            __syncthreads();

            // ---- kv tile: wave -> (r_half, n_tile); K = 128 tokens ---------
            {
                const int rh = wave >> 2;
                v8f acc = kvacc[rc];
                #pragma unroll
                for (int kk = 0; kk < 4; ++kk) {
                    v16h a = frag_ld(&kpT[(size_t)(rh * 16 + lo) * 136 +
                                          kk * 32 + khoff]);
                    acc = wmma16(a, bvf[kk], acc);
                }
                kvacc[rc] = acc;
            }
            // ---- ksum partial: thread -> (r = tid&31, 16-token group) ------
            {
                const int rl = tid & 31, grp = tid >> 5;
                float s = 0.f;
                #pragma unroll
                for (int t = 0; t < 16; ++t)
                    s += (float)kpT[(size_t)rl * 136 + grp * 16 + t];
                atomicAdd(&ksum_s[rc * 32 + rl], s);
            }
            __syncthreads();
        }

        if (ts + 1 < SUBS) wait_async0();   // next tile resident before reuse
        __syncthreads();
    }

    // ---- flush register kv accumulators (one atomic per output element) ---
    {
        const int rh = wave >> 2, nt = wave & 3;
        #pragma unroll
        for (int rc = 0; rc < RC_N; ++rc) {
            float* base = kv32 + ((size_t)head * RP + rc * 32 + rh * 16 + khoff) * D +
                          nt * 16 + lo;
            #pragma unroll
            for (int e = 0; e < 8; ++e) atomicAdd(base + (size_t)e * D, kvacc[rc][e]);
        }
    }
    for (int i = tid; i < RP; i += 256) atomicAdd(&ksum[head * RP + i], ksum_s[i]);
}

// ---------------------------------------------------------------------------
// Kernel 2: kv f32 [h][r][n] -> f16 transposed [h][n][r] for B-fragment loads
// ---------------------------------------------------------------------------
__global__ __launch_bounds__(256) void perf_mid(const float* __restrict__ kv32,
                                                _Float16* __restrict__ kv16T) {
    const int idx = blockIdx.x * 256 + threadIdx.x;
    if (idx >= HEADS * RP * D) return;
    const int h = idx / (RP * D);
    const int rem = idx % (RP * D);
    const int r = rem / D, n = rem % D;
    kv16T[((size_t)h * D + n) * RP + r] = (_Float16)kv32[idx];
}

// ---------------------------------------------------------------------------
// Kernel 3: qp = relu(q @ proj^T)+eps ; out = (qp @ kv) / (qp . ksum)
// grid (LSEQ/T2, HEADS) = (64, 32), 256 threads (8 waves)
// ---------------------------------------------------------------------------
__global__ __launch_bounds__(256) void perf_pass2(const float* __restrict__ Q,
                                                  const _Float16* __restrict__ proj16,
                                                  const float* __restrict__ ksum,
                                                  const _Float16* __restrict__ kv16T,
                                                  float* __restrict__ out) {
    __shared__ __attribute__((aligned(16))) float    stage_q[T2][D];  // 32 KB
    __shared__ __attribute__((aligned(16))) _Float16 qp[T2][40];      // 10 KB
    __shared__ float den[T2];

    const int tid   = threadIdx.x;
    const int wave  = tid >> 5;
    const int lane  = tid & 31;
    const int lo    = lane & 15;
    const int khoff = (lane >> 4) * 8;
    const int head  = blockIdx.y;
    const int tok0  = blockIdx.x * T2;
    const float* Qh = Q + (size_t)head * LSEQ * D;

    if (wave == 0) __builtin_prefetch(proj16 + (size_t)lane * 576, 0, 3);

    // async-stage the q tile (f32) into LDS
    for (int i = tid; i < T2 * D / 4; i += 256)
        async_b128(Qh + (size_t)tok0 * D + i * 4, &stage_q[0][0] + i * 4);
    if (tid < T2) den[tid] = 0.f;
    wait_async0();
    __syncthreads();

    // per-wave q A-fragments (M-tile = wave), converted once, reused 9x
    v16h aqf[2];
    {
        const float* row = &stage_q[wave * 16 + lo][0];
        #pragma unroll
        for (int kk = 0; kk < 2; ++kk) {
            v16h t;
            #pragma unroll
            for (int e = 0; e < 8; ++e) {
                t[e]     = (_Float16)row[kk * 32 + khoff + e];
                t[e + 8] = (_Float16)row[kk * 32 + khoff + 16 + e];
            }
            aqf[kk] = t;
        }
    }

    v8f oacc[4];
    #pragma unroll
    for (int nt = 0; nt < 4; ++nt) oacc[nt] = vzero8();

    for (int rc = 0; rc < RC_N; ++rc) {
        // ---- projection chunk + denominator contribution ------------------
        #pragma unroll
        for (int sub = 0; sub < 2; ++sub) {
            const int rt = rc * 2 + sub;
            v8f acc = vzero8();
            #pragma unroll
            for (int kk = 0; kk < 2; ++kk) {
                v16h b = frag_ld(proj16 + (size_t)(rt * 16 + lo) * D + kk * 32 + khoff);
                acc = wmma16(aqf[kk], b, acc);
            }
            const int   rglob  = rt * 16 + lo;
            const float addeps = (rglob < RNUM) ? EPSF : 0.f;
            const float sval   = ksum[head * RP + rglob];
            #pragma unroll
            for (int e = 0; e < 8; ++e) {
                float x = acc[e];
                x = (x > 0.f) ? x : 0.f;
                x += addeps;
                qp[wave * 16 + khoff + e][sub * 16 + lo] = (_Float16)x;
                atomicAdd(&den[wave * 16 + khoff + e], x * sval);   // ds_add_f32
            }
        }
        __syncthreads();

        // ---- out += qp_chunk @ kv_chunk (K = 32) --------------------------
        {
            v16h a = frag_ld(&qp[wave * 16 + lo][khoff]);
            #pragma unroll
            for (int nt = 0; nt < 4; ++nt) {
                v16h b = frag_ld(kv16T + ((size_t)head * D + nt * 16 + lo) * RP +
                                 rc * 32 + khoff);
                oacc[nt] = wmma16(a, b, oacc[nt]);
            }
        }
        __syncthreads();
    }

    float dinv[8];
    #pragma unroll
    for (int e = 0; e < 8; ++e) dinv[e] = 1.f / den[wave * 16 + khoff + e];

    #pragma unroll
    for (int nt = 0; nt < 4; ++nt) {
        #pragma unroll
        for (int e = 0; e < 8; ++e) {
            const int t = wave * 16 + khoff + e;
            out[((size_t)head * LSEQ + tok0 + t) * D + nt * 16 + lo] =
                oacc[nt][e] * dinv[e];
        }
    }
}

// ---------------------------------------------------------------------------
extern "C" void kernel_launch(void* const* d_in, const int* in_sizes, int n_in,
                              void* d_out, int out_size, void* d_ws, size_t ws_size,
                              hipStream_t stream) {
    (void)in_sizes; (void)n_in; (void)out_size; (void)ws_size;
    const float* q    = (const float*)d_in[0];
    const float* k    = (const float*)d_in[1];
    const float* v    = (const float*)d_in[2];
    const float* proj = (const float*)d_in[3];
    float* out = (float*)d_out;

    // workspace carve-up (total ~3.45 MB)
    char* ws = (char*)d_ws;
    _Float16* proj16 = (_Float16*)ws;                                   //  36,864 B
    float*    ksum   = (float*)(ws + 36864);                            //  36,864 B
    float*    kv32   = (float*)(ws + 73728);                            // 2,359,296 B
    _Float16* kv16T  = (_Float16*)(ws + 73728 + 2359296);               // 1,179,648 B

    const size_t smem1 = 4 * (size_t)T1 * D * sizeof(float)             // 131,072
                       + 32 * 136 * sizeof(_Float16)                    //   8,704
                       + RP * sizeof(float);                            //   1,152

    const int initBlocks = (HEADS * RP * D + 255) / 256;                // 2304
    perf_init<<<initBlocks, 256, 0, stream>>>(proj, proj16, ksum, kv32);
    perf_pass1<<<dim3(LSEQ / (T1 * SUBS), HEADS), 256, smem1, stream>>>(k, v, proj16,
                                                                        ksum, kv32);
    perf_mid<<<initBlocks, 256, 0, stream>>>(kv32, kv16T);
    perf_pass2<<<dim3(LSEQ / T2, HEADS), 256, 0, stream>>>(q, proj16, ksum,
                                                           kv16T, out);
}
